// BitLinear_15848429322573
// MI455X (gfx1250) — compile-verified, matrix operand order checked by reading, not run on hardware
//
#include <hip/hip_runtime.h>
#include <stdint.h>

#define OUT_F 11008
#define IN_F  4096
#define TOKENS 1024
#define KP 256                  // K-panel staged in LDS per iteration
#define LDS_STRIDE (KP + 8)     // ushort elements; +8 pad (16B) staggers banks

typedef __bf16 v16bf __attribute__((ext_vector_type(16)));
typedef float  v8f   __attribute__((ext_vector_type(8)));

union BF16Frag { v16bf v; unsigned u[8]; };

// fp32 -> bf16 round-to-nearest-even
__device__ __forceinline__ unsigned short f2bf(float f) {
  unsigned u = __float_as_uint(f);
  return (unsigned short)((u + 0x7FFFu + ((u >> 16) & 1u)) >> 16);
}

// nbyte = ~packed_byte. Element j sign bit = bit (7-j) of nbyte (bit==1 -> +1).
// Builds bf16 pair for elements j0 (low 16) / j0+1 (high 16):
//   bit (7-j0)   -> position 15  : shift left by (8 + j0)
//   bit (6-j0)   -> position 31  : shift left by (25 + j0)
__device__ __forceinline__ unsigned pack_pm1(unsigned nbyte, int j0) {
  return 0x3F803F80u | ((nbyte << (8 + j0)) & 0x8000u)
                     | ((nbyte << (25 + j0)) & 0x80000000u);
}

__global__ void __launch_bounds__(256)
cvt_bf16(const float* __restrict__ x, unsigned short* __restrict__ o) {
  size_t i = ((size_t)blockIdx.x * blockDim.x + threadIdx.x) * 4;
  float4 f = *(const float4*)(x + i);
  unsigned lo = (unsigned)f2bf(f.x) | ((unsigned)f2bf(f.y) << 16);
  unsigned hi = (unsigned)f2bf(f.z) | ((unsigned)f2bf(f.w) << 16);
  *(uint2*)(o + i) = make_uint2(lo, hi);
}

// Block = 8 waves. Block tile: M=128 x N=128. Wave w: columns n0..n0+15,
// 8 M-subtiles (one unpacked B fragment feeds 8 WMMAs).
// X K-panel (128 x KP bf16, 66KB) staged in LDS, shared by all waves.
template<bool USE_WS>
__global__ void __launch_bounds__(256)
bitlinear_wmma(const float* __restrict__ x, const unsigned short* __restrict__ xbf,
               const int* __restrict__ bp, float* __restrict__ out)
{
  __shared__ unsigned short As[128 * LDS_STRIDE];

  const int lane = threadIdx.x & 31;
  const int wave = threadIdx.x >> 5;
  const int h    = lane >> 4;      // lane half (selects K sub-range per WMMA layout)
  const int l    = lane & 15;
  const int n0   = blockIdx.x * 128 + wave * 16;
  const int m0   = blockIdx.y * 128;

  // Packed-weight row for this lane's output column (512 int32 "bytes" per row)
  const int* brow = bp + (size_t)(n0 + l) * (IN_F / 8);

  v8f acc[8] = {};

  // Cooperative staging: 256 threads cover 128 rows x KP cols (2 threads/row)
  const int sr = threadIdx.x >> 1;   // row 0..127
  const int sq = threadIdx.x & 1;    // half of the row (128 ushorts = 256B)

  int4 e = *(const int4*)brow;       // packed bits for k = 0..31 (double-buffered)

  for (int kk = 0; kk < IN_F; kk += KP) {
    __syncthreads();
    unsigned short* dst = &As[sr * LDS_STRIDE + sq * 128];
    if (USE_WS) {
      const unsigned short* src = xbf + (size_t)(m0 + sr) * IN_F + kk + sq * 128;
      #pragma unroll
      for (int i = 0; i < 16; ++i)
        ((uint4*)dst)[i] = ((const uint4*)src)[i];
      __builtin_prefetch(src + KP, 0, 1);   // speculative: dropped if OOB
    } else {
      const float* src = x + (size_t)(m0 + sr) * IN_F + kk + sq * 128;
      #pragma unroll
      for (int i = 0; i < 16; ++i) {
        float4 f0 = ((const float4*)src)[2 * i];
        float4 f1 = ((const float4*)src)[2 * i + 1];
        uint4 p;
        p.x = (unsigned)f2bf(f0.x) | ((unsigned)f2bf(f0.y) << 16);
        p.y = (unsigned)f2bf(f0.z) | ((unsigned)f2bf(f0.w) << 16);
        p.z = (unsigned)f2bf(f1.x) | ((unsigned)f2bf(f1.y) << 16);
        p.w = (unsigned)f2bf(f1.z) | ((unsigned)f2bf(f1.w) << 16);
        ((uint4*)dst)[i] = p;
      }
      __builtin_prefetch(src + KP, 0, 1);
    }
    __syncthreads();

    #pragma unroll
    for (int k0 = 0; k0 < KP; k0 += 32) {
      // Preload next k-step's packed bits (clamped to stay in-bounds; the
      // wrapped value is never consumed on the final iteration).
      int gn = kk + k0 + 32;
      gn = (gn < IN_F) ? gn : 0;
      int4 en = *(const int4*)(brow + (gn >> 3));

      // Unpack current 32 bits -> B fragment (lane = column n, K split by h)
      unsigned na = ~(unsigned)(h ? e.y : e.x);   // K = 8h .. 8h+7   (local)
      unsigned nb = ~(unsigned)(h ? e.w : e.z);   // K = 16+8h .. 16+8h+7
      BF16Frag B;
      #pragma unroll
      for (int d = 0; d < 4; ++d) {
        B.u[d]     = pack_pm1(na, 2 * d);
        B.u[4 + d] = pack_pm1(nb, 2 * d);
      }

      // 8 WMMAs with A-fragment double buffering (load t+1 before WMMA t)
      BF16Frag A[2];
      const unsigned short* abase = &As[l * LDS_STRIDE + k0 + 8 * h];
      *(uint4*)&A[0].u[0] = *(const uint4*)abase;          // K = 8h..8h+7
      *(uint4*)&A[0].u[4] = *(const uint4*)(abase + 16);   // K = 16+8h..+7
      #pragma unroll
      for (int t = 0; t < 8; ++t) {
        if (t < 7) {
          const unsigned short* ap = abase + (t + 1) * 16 * LDS_STRIDE;
          *(uint4*)&A[(t + 1) & 1].u[0] = *(const uint4*)ap;
          *(uint4*)&A[(t + 1) & 1].u[4] = *(const uint4*)(ap + 16);
        }
        acc[t] = __builtin_amdgcn_wmma_f32_16x16x32_bf16(
            false, A[t & 1].v, false, B.v, (short)0, acc[t], false, false);
      }
      e = en;
    }
  }

  // C/D layout: VGPR r of lane -> row (r + 8h), col l
  #pragma unroll
  for (int t = 0; t < 8; ++t) {
    float* op = out + (size_t)(m0 + t * 16 + 8 * h) * OUT_F + n0 + l;
    #pragma unroll
    for (int r = 0; r < 8; ++r)
      op[(size_t)r * OUT_F] = acc[t][r];
  }
}

extern "C" void kernel_launch(void* const* d_in, const int* in_sizes, int n_in,
                              void* d_out, int out_size, void* d_ws, size_t ws_size,
                              hipStream_t stream) {
  const float* x  = (const float*)d_in[0];
  const int*   bp = (const int*)d_in[1];
  float* out = (float*)d_out;

  const size_t need = (size_t)TOKENS * IN_F * sizeof(unsigned short);
  dim3 grid(OUT_F / 128, TOKENS / 128);   // 86 x 8 blocks, 8 waves each

  if (d_ws && ws_size >= need) {
    unsigned short* xbf = (unsigned short*)d_ws;
    cvt_bf16<<<(TOKENS * IN_F / 4) / 256, 256, 0, stream>>>(x, xbf);
    bitlinear_wmma<true><<<grid, 256, 0, stream>>>(x, xbf, bp, out);
  } else {
    bitlinear_wmma<false><<<grid, 256, 0, stream>>>(x, nullptr, bp, out);
  }
}
